// PyGPointNextDecoder_60610578481732
// MI455X (gfx1250) — matrix operand order repaired and checked by reading, not compiled, not run on hardware
//
#include <hip/hip_runtime.h>
#include <hip/hip_bf16.h>

typedef __attribute__((ext_vector_type(16))) _Float16 v16h;
typedef __attribute__((ext_vector_type(8)))  _Float16 v8h;
typedef __attribute__((ext_vector_type(8)))  float    v8f;

#ifndef FLT_BIG
#define FLT_BIG 3.4e38f
#endif

// ---------------------------------------------------------------------------
// kNN (k=3) per fine point, equal-sized sorted batches (B=8).
// One thread per fine point; register top-3 insertion (ties keep lower index,
// matching jax.lax.top_k). Writes indices and normalized inverse-d2 weights.
// ---------------------------------------------------------------------------
__global__ void knn3_kernel(const float* __restrict__ pf, const float* __restrict__ pc,
                            int n1, int n2, int nb,
                            int* __restrict__ knn_idx, float* __restrict__ knn_wn) {
  int i = blockIdx.x * blockDim.x + threadIdx.x;
  if (i >= n1) return;
  int n1pb = n1 / nb, n2pb = n2 / nb;
  int b = i / n1pb;
  float px = pf[3 * i + 0], py = pf[3 * i + 1], pz = pf[3 * i + 2];
  float d0 = FLT_BIG, d1 = FLT_BIG, d2 = FLT_BIG;
  int   i0 = 0, i1 = 0, i2 = 0;
  int jb = b * n2pb;
  for (int j = 0; j < n2pb; ++j) {
    const float* q = pc + 3 * (size_t)(jb + j);
    float dx = px - q[0], dy = py - q[1], dz = pz - q[2];
    float d = dx * dx + dy * dy + dz * dz;
    if (d < d0)      { d2 = d1; i2 = i1; d1 = d0; i1 = i0; d0 = d; i0 = jb + j; }
    else if (d < d1) { d2 = d1; i2 = i1; d1 = d;  i1 = jb + j; }
    else if (d < d2) { d2 = d;  i2 = jb + j; }
  }
  float w0 = 1.0f / fmaxf(d0, 1e-16f);
  float w1 = 1.0f / fmaxf(d1, 1e-16f);
  float w2 = 1.0f / fmaxf(d2, 1e-16f);
  float ws = w0 + w1 + w2;
  knn_idx[3 * i + 0] = i0;  knn_idx[3 * i + 1] = i1;  knn_idx[3 * i + 2] = i2;
  knn_wn[3 * i + 0] = w0 / ws;  knn_wn[3 * i + 1] = w1 / ws;  knn_wn[3 * i + 2] = w2 / ws;
}

// ---------------------------------------------------------------------------
// Interpolate coarse features + concat with fine features; emit f16 GEMM input
// x = [f_fine | knn_interp(f_coarse)]  -> xh[n1, C1+C2] (f16)
// ---------------------------------------------------------------------------
__global__ void interp_concat_kernel(const float* __restrict__ ff, const float* __restrict__ fc,
                                     const int* __restrict__ knn_idx, const float* __restrict__ knn_wn,
                                     _Float16* __restrict__ xh, int n1, int C1, int C2) {
  int Cin = C1 + C2;
  size_t t = (size_t)blockIdx.x * blockDim.x + threadIdx.x;
  if (t >= (size_t)n1 * Cin) return;
  int i = (int)(t / Cin);
  int c = (int)(t % Cin);
  float v;
  if (c < C1) {
    v = ff[(size_t)i * C1 + c];
  } else {
    int c2 = c - C1;
    v = knn_wn[3 * i + 0] * fc[(size_t)knn_idx[3 * i + 0] * C2 + c2]
      + knn_wn[3 * i + 1] * fc[(size_t)knn_idx[3 * i + 1] * C2 + c2]
      + knn_wn[3 * i + 2] * fc[(size_t)knn_idx[3 * i + 2] * C2 + c2];
  }
  xh[t] = (_Float16)v;
}

// ---------------------------------------------------------------------------
// Weight convert + transpose: W[Cin,Ch] f32 -> Wt[Ch,Cin] f16.
// Transposing makes each lane's B fragment two contiguous 16-byte runs.
// ---------------------------------------------------------------------------
__global__ void wt_f16_kernel(const float* __restrict__ W, _Float16* __restrict__ Wt,
                              int Cin, int Ch) {
  int t = blockIdx.x * blockDim.x + threadIdx.x;
  if (t >= Cin * Ch) return;
  int k = t / Ch, c = t % Ch;
  Wt[(size_t)c * Cin + k] = (_Float16)W[t];
}

// ---------------------------------------------------------------------------
// WMMA GEMM: H[N,Ch] = Xh[N,Cin](f16) * Wt[Ch,Cin](f16, pre-transposed) + bias.
// One 16x16 output tile per wave32; 8 waves per block -> 128 rows per block.
// Per K-step of 32, each lane issues 4x global_load_b128 + 1 v_wmma:
//   A frag: row = lane%16;  a[0..7]  = X[row][k0+8h .. +7]   (h = lane/16)
//                           a[8..15] = X[row][k0+16+8h .. +7]
//   B frag: col = lane%16;  b[0..7]  = Wt[col][k0+8h .. +7], b[8..15] = Wt[col][k0+16+8h .. +7]
//   C/D   : M = r + 8h, N = lane%16  (8 f32 VGPRs)
// Cin is always a multiple of 32; N a multiple of 128; Ch a multiple of 16.
// All fragment loads are 16-byte aligned (buffers 256-B aligned).
// ---------------------------------------------------------------------------
__global__ void gemm_wmma_kernel(const _Float16* __restrict__ X, const _Float16* __restrict__ Wt,
                                 const float* __restrict__ bias, float* __restrict__ H,
                                 int N, int Cin, int Ch) {
  int lane = threadIdx.x;                 // 0..31
  int wave = threadIdx.y;                 // 0..7
  int rowBase = (blockIdx.y * 8 + wave) * 16;
  int colBase = blockIdx.x * 16;
  if (rowBase >= N) return;               // wave-uniform: EXEC stays all-ones
  int half = lane >> 4;
  int l16  = lane & 15;

  const _Float16* xp = X  + (size_t)(rowBase + l16) * Cin + half * 8;
  const _Float16* bp = Wt + (size_t)(colBase + l16) * Cin + half * 8;

  v8f acc = {};
  for (int k0 = 0; k0 < Cin; k0 += 32) {
    __builtin_prefetch(xp + k0 + 32, 0, 1);   // global_prefetch_b8 next K tile
    v8h a0 = *(const v8h*)(xp + k0);
    v8h a1 = *(const v8h*)(xp + k0 + 16);
    v8h b0 = *(const v8h*)(bp + k0);
    v8h b1 = *(const v8h*)(bp + k0 + 16);
    v16h a = __builtin_shufflevector(a0, a1, 0, 1, 2, 3, 4, 5, 6, 7,
                                     8, 9, 10, 11, 12, 13, 14, 15);
    v16h b = __builtin_shufflevector(b0, b1, 0, 1, 2, 3, 4, 5, 6, 7,
                                     8, 9, 10, 11, 12, 13, 14, 15);
    acc = __builtin_amdgcn_wmma_f32_16x16x32_f16(
        /*neg_a=*/false, a, /*neg_b=*/false, b,
        /*c_mod=*/(short)0, acc, /*reuse_a=*/false, /*reuse_b=*/false);
  }
  float bcol = bias[colBase + l16];
#pragma unroll
  for (int r = 0; r < 8; ++r) {
    int row = rowBase + r + 8 * half;
    H[(size_t)row * Ch + colBase + l16] = acc[r] + bcol;
  }
}

// ---------------------------------------------------------------------------
// Batch-norm stats over the N rows of one channel; fold gamma/beta into
// scale/shift:  y = h*scale[c] + shift[c]
// ---------------------------------------------------------------------------
__global__ void bn_stats_kernel(const float* __restrict__ H, int N, int Ch,
                                const float* __restrict__ gamma, const float* __restrict__ beta,
                                float* __restrict__ scale, float* __restrict__ shift) {
  int c = blockIdx.x;
  __shared__ float s1[256];
  __shared__ float s2[256];
  float a = 0.f, b = 0.f;
  for (int i = threadIdx.x; i < N; i += blockDim.x) {
    float v = H[(size_t)i * Ch + c];
    a += v; b += v * v;
  }
  s1[threadIdx.x] = a; s2[threadIdx.x] = b;
  __syncthreads();
  for (int st = 128; st > 0; st >>= 1) {
    if ((int)threadIdx.x < st) {
      s1[threadIdx.x] += s1[threadIdx.x + st];
      s2[threadIdx.x] += s2[threadIdx.x + st];
    }
    __syncthreads();
  }
  if (threadIdx.x == 0) {
    float inv_n = 1.0f / (float)N;
    float mu  = s1[0] * inv_n;
    float var = s2[0] * inv_n - mu * mu;
    float sc  = gamma[c] * rsqrtf(var + 1e-5f);
    scale[c] = sc;
    shift[c] = beta[c] - mu * sc;
  }
}

// ---------------------------------------------------------------------------
// Apply BN + ReLU. Optionally emit f32 (stage output) and/or f16 (next GEMM).
// ---------------------------------------------------------------------------
__global__ void bn_relu_kernel(const float* __restrict__ H,
                               const float* __restrict__ scale, const float* __restrict__ shift,
                               float* __restrict__ outF, _Float16* __restrict__ outH,
                               int N, int Ch) {
  size_t t = (size_t)blockIdx.x * blockDim.x + threadIdx.x;
  if (t >= (size_t)N * Ch) return;
  int c = (int)(t % Ch);
  float v = fmaxf(H[t] * scale[c] + shift[c], 0.0f);
  if (outF) outF[t] = v;
  if (outH) outH[t] = (_Float16)v;
}

// ---------------------------------------------------------------------------
// Host orchestration
// ---------------------------------------------------------------------------
extern "C" void kernel_launch(void* const* d_in, const int* in_sizes, int n_in,
                              void* d_out, int out_size, void* d_ws, size_t ws_size,
                              hipStream_t stream) {
  (void)in_sizes; (void)n_in; (void)out_size; (void)ws_size;
  const float* p[5]; const float* f[5];
  for (int i = 0; i < 5; ++i) {
    p[i] = (const float*)d_in[3 * i + 0];
    f[i] = (const float*)d_in[3 * i + 1];
    // b{i} (batch vectors) unused: batches are equal-sized and sorted.
  }

  // Workspace bump allocator (max footprint ~33 MB).
  char* wsp = (char*)d_ws;
  auto alloc = [&](size_t bytes) -> void* {
    void* r = (void*)wsp;
    wsp += (bytes + 255) & ~(size_t)255;
    return r;
  };
  float*    fb3     = (float*)alloc((size_t)1024 * 256 * 4);   // stage-a output
  float*    fb2     = (float*)alloc((size_t)4096 * 128 * 4);   // stage-b output
  float*    fb1     = (float*)alloc((size_t)16384 * 64 * 4);   // stage-c output
  int*      knn_idx = (int*)alloc((size_t)65536 * 3 * 4);
  float*    knn_wn  = (float*)alloc((size_t)65536 * 3 * 4);
  _Float16* xh      = (_Float16*)alloc((size_t)65536 * 96 * 2);  // max N*Cin
  _Float16* yh      = (_Float16*)alloc((size_t)65536 * 32 * 2);  // max N*Ch
  float*    hbuf    = (float*)alloc((size_t)65536 * 32 * 4);     // max N*Ch
  _Float16* wt1     = (_Float16*)alloc((size_t)768 * 256 * 2);   // W1^T f16
  _Float16* wt2     = (_Float16*)alloc((size_t)256 * 256 * 2);   // W2^T f16
  float*    scl     = (float*)alloc(512 * 4);
  float*    shf     = (float*)alloc(512 * 4);

  auto run_stage = [&](const float* pf, const float* ff, int n1, int C1,
                       const float* pc, const float* fc, int n2, int C2,
                       int pbase, float* outF) {
    const float* w1  = (const float*)d_in[pbase + 0];
    const float* b1  = (const float*)d_in[pbase + 1];
    const float* g1  = (const float*)d_in[pbase + 2];
    const float* bt1 = (const float*)d_in[pbase + 3];
    const float* w2  = (const float*)d_in[pbase + 4];
    const float* b2  = (const float*)d_in[pbase + 5];
    const float* g2  = (const float*)d_in[pbase + 6];
    const float* bt2 = (const float*)d_in[pbase + 7];
    int Cin = C1 + C2;

    knn3_kernel<<<(n1 + 255) / 256, 256, 0, stream>>>(pf, pc, n1, n2, 8, knn_idx, knn_wn);

    size_t tot = (size_t)n1 * Cin;
    interp_concat_kernel<<<(unsigned)((tot + 255) / 256), 256, 0, stream>>>(
        ff, fc, knn_idx, knn_wn, xh, n1, C1, C2);

    wt_f16_kernel<<<(Cin * C1 + 255) / 256, 256, 0, stream>>>(w1, wt1, Cin, C1);
    wt_f16_kernel<<<(C1 * C1 + 255) / 256, 256, 0, stream>>>(w2, wt2, C1, C1);

    dim3 gblk(32, 8);
    // Linear1 + BN + ReLU  (emit f16 input for Linear2)
    gemm_wmma_kernel<<<dim3(C1 / 16, n1 / 128), gblk, 0, stream>>>(xh, wt1, b1, hbuf, n1, Cin, C1);
    bn_stats_kernel<<<C1, 256, 0, stream>>>(hbuf, n1, C1, g1, bt1, scl, shf);
    bn_relu_kernel<<<(unsigned)(((size_t)n1 * C1 + 255) / 256), 256, 0, stream>>>(
        hbuf, scl, shf, nullptr, yh, n1, C1);
    // Linear2 + BN + ReLU  (emit f32 stage output)
    gemm_wmma_kernel<<<dim3(C1 / 16, n1 / 128), gblk, 0, stream>>>(yh, wt2, b2, hbuf, n1, C1, C1);
    bn_stats_kernel<<<C1, 256, 0, stream>>>(hbuf, n1, C1, g2, bt2, scl, shf);
    bn_relu_kernel<<<(unsigned)(((size_t)n1 * C1 + 255) / 256), 256, 0, stream>>>(
        hbuf, scl, shf, outF, nullptr, n1, C1);
  };

  // Stage a: fine scale-3 (1024, 256ch) <- coarse scale-4 (256, 512ch)
  run_stage(p[3], f[3], 1024, 256, p[4], f[4], 256, 512, 15, fb3);
  // Stage b: fine scale-2 (4096, 128ch) <- coarse = stage-a output
  run_stage(p[2], f[2], 4096, 128, p[3], fb3, 1024, 256, 23, fb2);
  // Stage c: fine scale-1 (16384, 64ch) <- coarse = stage-b output
  run_stage(p[1], f[1], 16384, 64, p[2], fb2, 4096, 128, 31, fb1);
  // Stage d: fine scale-0 (65536, 32ch) <- coarse = stage-c output -> d_out
  run_stage(p[0], f[0], 65536, 32, p[1], fb1, 16384, 64, 39, (float*)d_out);
}
